// CausalWanSelfAttention_61641370632295
// MI455X (gfx1250) — matrix operand order repaired and checked by reading, not compiled, not run on hardware
//
#include <hip/hip_runtime.h>
#include <hip/hip_bf16.h>
#include <math.h>

#define SEQ   3072
#define DIMN  1536
#define NHEAD 12
#define HDIM  128

typedef __attribute__((ext_vector_type(16))) __bf16 v16bf;
typedef __attribute__((ext_vector_type(8)))  __bf16 v8bf;
typedef __attribute__((ext_vector_type(4)))  __bf16 v4bf;
typedef __attribute__((ext_vector_type(8)))  float  v8f;

static __device__ __forceinline__ v8f wmma_bf16(v16bf a, v16bf b, v8f c) {
  return __builtin_amdgcn_wmma_f32_16x16x32_bf16(false, a, false, b, (short)0, c, false, false);
}

// A-operand (16x32 bf16): lane r=L%16,g=L/16; halves 0-15 = K g*16..g*16+15 (contiguous)
static __device__ __forceinline__ v16bf load_frag_a(const __bf16* p) {
  v8bf lo = *(const v8bf*)p;
  v8bf hi = *(const v8bf*)(p + 8);
  return __builtin_shufflevector(lo, hi, 0,1,2,3,4,5,6,7,8,9,10,11,12,13,14,15);
}
// B-operand (32x16 bf16): lane n=L%16,g=L/16; halves 0-15 = K g*16..g*16+15 (contiguous)
static __device__ __forceinline__ v16bf load_frag_b(const __bf16* p) {
  v8bf lo = *(const v8bf*)p;
  v8bf hi = *(const v8bf*)(p + 8);
  return __builtin_shufflevector(lo, hi, 0,1,2,3,4,5,6,7,8,9,10,11,12,13,14,15);
}

// ---------------- cast f32 -> bf16 (vectorized) ----------------
__global__ void k_cast_bf16(const float* __restrict__ in, __bf16* __restrict__ out, int n4) {
  int i = blockIdx.x * blockDim.x + threadIdx.x;
  if (i < n4) {
    float4 v = ((const float4*)in)[i];
    v4bf o = { (__bf16)v.x, (__bf16)v.y, (__bf16)v.z, (__bf16)v.w };
    ((v4bf*)out)[i] = o;
  }
}

// ---------------- GEMM: out[M,N] = A[M,K] @ B[N,K]^T + bias ----------------
// 4 waves/block, each wave computes a 16x64 tile with v_wmma_f32_16x16x32_bf16.
template<bool OUT_F32>
__global__ __launch_bounds__(128) void
k_gemm(const __bf16* __restrict__ A, const __bf16* __restrict__ B,
       const float* __restrict__ bias, void* __restrict__ out,
       int M, int N, int K) {
  const int lane = threadIdx.x & 31;
  const int wave = threadIdx.x >> 5;
  const int r = lane & 15, g = lane >> 4;
  const int m_base = blockIdx.y * 16;
  const int n_base = blockIdx.x * 256 + wave * 64;

  const __bf16* ap = A + (size_t)(m_base + r) * K + g * 16;
  const __bf16* bp = B + (size_t)(n_base + r) * K + g * 16;

  v8f c[4] = {};
  for (int kk = 0; kk < K; kk += 32) {
    v16bf a = load_frag_a(ap + kk);
#pragma unroll
    for (int t = 0; t < 4; ++t) {
      v16bf b = load_frag_b(bp + (size_t)t * 16 * K + kk);
      c[t] = wmma_bf16(a, b, c[t]);
    }
  }
#pragma unroll
  for (int t = 0; t < 4; ++t) {
    int col = n_base + t * 16 + r;
    float bv = bias[col];
#pragma unroll
    for (int j = 0; j < 8; ++j) {
      int row = m_base + j + 8 * g;
      float v = c[t][j] + bv;
      if (OUT_F32) ((float*)out)[(size_t)row * N + col] = v;
      else         ((__bf16*)out)[(size_t)row * N + col] = (__bf16)v;
    }
  }
}

// ---------------- RMSNorm + RoPE + head reshape (+ V transpose) ----------------
// one block (256 thr) per token
__global__ void k_normrope(const __bf16* __restrict__ qf, const __bf16* __restrict__ kf,
                           const __bf16* __restrict__ vf,
                           const float* __restrict__ gq, const float* __restrict__ gk,
                           const float* __restrict__ fcos, const float* __restrict__ fsin,
                           __bf16* __restrict__ qh, __bf16* __restrict__ kh,
                           __bf16* __restrict__ vt) {
  const int tok = blockIdx.x;
  const int t = threadIdx.x;
  const __bf16* qrow = qf + (size_t)tok * DIMN;
  const __bf16* krow = kf + (size_t)tok * DIMN;
  const __bf16* vrow = vf + (size_t)tok * DIMN;

  float sq = 0.f, sk = 0.f;
  for (int i = t; i < DIMN; i += 256) {
    float a = (float)qrow[i]; sq += a * a;
    float b = (float)krow[i]; sk += b * b;
  }
  __shared__ float sA[256], sB[256];
  sA[t] = sq; sB[t] = sk;
  __syncthreads();
  for (int s = 128; s > 0; s >>= 1) {
    if (t < s) { sA[t] += sA[t + s]; sB[t] += sB[t + s]; }
    __syncthreads();
  }
  const float rq = rsqrtf(sA[0] * (1.0f / DIMN) + 1e-6f);
  const float rk = rsqrtf(sB[0] * (1.0f / DIMN) + 1e-6f);

  const int f  = tok >> 8;
  const int hh = (tok >> 4) & 15;
  const int ww = tok & 15;

  for (int p = t; p < (DIMN / 2); p += 256) {   // 768 rope pairs
    int head = p >> 6;
    int c = p & 63;
    int d0 = head * HDIM + 2 * c;
    int prow = (c < 22) ? f : ((c < 43) ? hh : ww);
    float cs = fcos[prow * 64 + c];
    float sn = fsin[prow * 64 + c];

    float xr = (float)qrow[d0] * rq * gq[d0];
    float xi = (float)qrow[d0 + 1] * rq * gq[d0 + 1];
    __bf16* qo = qh + ((size_t)head * SEQ + tok) * HDIM + 2 * c;
    qo[0] = (__bf16)(xr * cs - xi * sn);
    qo[1] = (__bf16)(xr * sn + xi * cs);

    float yr = (float)krow[d0] * rk * gk[d0];
    float yi = (float)krow[d0 + 1] * rk * gk[d0 + 1];
    __bf16* ko = kh + ((size_t)head * SEQ + tok) * HDIM + 2 * c;
    ko[0] = (__bf16)(yr * cs - yi * sn);
    ko[1] = (__bf16)(yr * sn + yi * cs);
  }
  // V: [S][DIM] -> [NH][HD][S]
  for (int i = t; i < DIMN; i += 256) {
    int head = i >> 7, hd = i & 127;
    vt[((size_t)head * HDIM + hd) * SEQ + tok] = vrow[i];
  }
}

// full-tile running-max butterfly across all 32 lanes via immediate ds_swizzle
static __device__ __forceinline__ float tile_max32(float x) {
  x = fmaxf(x, __int_as_float(__builtin_amdgcn_ds_swizzle(__float_as_int(x), 0x041F)));
  x = fmaxf(x, __int_as_float(__builtin_amdgcn_ds_swizzle(__float_as_int(x), 0x081F)));
  x = fmaxf(x, __int_as_float(__builtin_amdgcn_ds_swizzle(__float_as_int(x), 0x101F)));
  x = fmaxf(x, __int_as_float(__builtin_amdgcn_ds_swizzle(__float_as_int(x), 0x201F)));
  x = fmaxf(x, __int_as_float(__builtin_amdgcn_ds_swizzle(__float_as_int(x), 0x401F)));
  return x;
}

// ---------------- Flash attention, block-causal + sink (dense per q-tile) ----------------
// grid(48, 12), 128 threads = 4 waves; each wave owns one 16-query tile.
__global__ __launch_bounds__(128, 1) void
k_attn(const __bf16* __restrict__ qh, const __bf16* __restrict__ kh,
       const __bf16* __restrict__ vt, __bf16* __restrict__ ab) {
  const int lane = threadIdx.x & 31;
  const int wave = threadIdx.x >> 5;
  const int head = blockIdx.y;
  const int qtile = blockIdx.x * 4 + wave;
  const int r = lane & 15, g = lane >> 4;
  const int q0 = qtile * 16;

  __shared__ __attribute__((aligned(16))) __bf16 lds[4][16][40]; // padded 16x32 P tile per wave

  // Q fragments (16 rows x 128 hd) for all 4 K-steps
  const __bf16* qp = qh + ((size_t)head * SEQ + q0 + r) * HDIM + g * 16;
  v16bf qa[4];
#pragma unroll
  for (int i = 0; i < 4; ++i) qa[i] = load_frag_a(qp + i * 32);

  v16bf ones;
#pragma unroll
  for (int i = 0; i < 16; ++i) ones[i] = (__bf16)1.0f;

  v8f o[8] = {};
  v8f lacc = {};          // running row-sums, replicated across columns (P @ ones)
  float m = -INFINITY;    // running tile max (exact online softmax with shared max)

  const float sc = 0.08838834764831845f * 1.4426950408889634f; // 1/sqrt(128) * log2(e)

  const int qb = qtile / 48;     // 768-token block index of this q tile
  int starts[2], lens[2]; int nr;
  if (qb <= 1) { starts[0] = 0; lens[0] = (qb + 1) * 768; nr = 1; }
  else { starts[0] = 0; lens[0] = 256; starts[1] = (qb - 1) * 768; lens[1] = 1536; nr = 2; }

  for (int ri = 0; ri < nr; ++ri) {
    const int kend = starts[ri] + lens[ri];
    for (int kv = starts[ri]; kv < kend; kv += 32) {
      // ---- scores: two 16x16 tiles over 32 keys, K-reduction over 128 hd ----
      v8f s0 = {}, s1 = {};
      const __bf16* kp0 = kh + ((size_t)head * SEQ + kv + r) * HDIM + g * 16;
      const __bf16* kp1 = kp0 + (size_t)16 * HDIM;
#pragma unroll
      for (int i = 0; i < 4; ++i) {
        v16bf b0 = load_frag_b(kp0 + i * 32);
        v16bf b1 = load_frag_b(kp1 + i * 32);
        s0 = wmma_bf16(qa[i], b0, s0);
        s1 = wmma_bf16(qa[i], b1, s1);
      }
      // ---- online softmax with per-tile running max (no masking: ranges exact) ----
      float a0[8], a1[8];
      float tmax = -INFINITY;
#pragma unroll
      for (int j = 0; j < 8; ++j) {
        a0[j] = s0[j] * sc;
        a1[j] = s1[j] * sc;
        tmax = fmaxf(tmax, fmaxf(a0[j], a1[j]));
      }
      tmax = tile_max32(tmax);
      const float mn = fmaxf(m, tmax);
      const float al = exp2f(m - mn);
      m = mn;
      // exp + write P (C-layout) into LDS in one pass
#pragma unroll
      for (int j = 0; j < 8; ++j) {
        lds[wave][j + 8 * g][r]      = (__bf16)exp2f(a0[j] - mn);
        lds[wave][j + 8 * g][16 + r] = (__bf16)exp2f(a1[j] - mn);
      }
      // rescale accumulators (uniform alpha)
#pragma unroll
      for (int t = 0; t < 8; ++t)
#pragma unroll
        for (int j = 0; j < 8; ++j) o[t][j] *= al;
#pragma unroll
      for (int j = 0; j < 8; ++j) lacc[j] *= al;

      asm volatile("s_wait_dscnt 0" ::: "memory");   // wave-local LDS RAW
      v16bf pa = load_frag_a(&lds[wave][r][g * 16]);

      // ---- O += P @ V (8 n-tiles), l += P @ ones ----
      const __bf16* vp = vt + ((size_t)head * HDIM + r) * SEQ + kv + g * 16;
#pragma unroll
      for (int t = 0; t < 8; ++t) {
        v16bf vb = load_frag_b(vp + (size_t)t * 16 * SEQ);
        o[t] = wmma_bf16(pa, vb, o[t]);
      }
      lacc = wmma_bf16(pa, ones, lacc);
    }
  }

  // ---- epilogue: normalize and scatter into [S][DIM] bf16 ----
#pragma unroll
  for (int t = 0; t < 8; ++t) {
    int col = head * HDIM + t * 16 + r;
#pragma unroll
    for (int j = 0; j < 8; ++j) {
      int row = q0 + j + 8 * g;
      ab[(size_t)row * DIMN + col] = (__bf16)(o[t][j] / lacc[j]);
    }
  }
}

// ---------------- launch ----------------
extern "C" void kernel_launch(void* const* d_in, const int* in_sizes, int n_in,
                              void* d_out, int out_size, void* d_ws, size_t ws_size,
                              hipStream_t stream) {
  (void)in_sizes; (void)n_in; (void)out_size; (void)ws_size;
  const float* x  = (const float*)d_in[0];
  const float* Wq = (const float*)d_in[1];
  const float* bq = (const float*)d_in[2];
  const float* Wk = (const float*)d_in[3];
  const float* bk = (const float*)d_in[4];
  const float* Wv = (const float*)d_in[5];
  const float* bv = (const float*)d_in[6];
  const float* Wo = (const float*)d_in[7];
  const float* bo = (const float*)d_in[8];
  const float* gq = (const float*)d_in[9];
  const float* gk = (const float*)d_in[10];
  const float* fc = (const float*)d_in[11];
  const float* fs = (const float*)d_in[12];

  char* ws = (char*)d_ws;
  const size_t SZ_X = (size_t)SEQ * DIMN * 2;   // 9,437,184 B
  const size_t SZ_W = (size_t)DIMN * DIMN * 2;  // 4,718,592 B
  __bf16* xb  = (__bf16*)(ws);
  __bf16* wqb = (__bf16*)(ws + SZ_X);
  __bf16* wkb = (__bf16*)(ws + SZ_X + SZ_W);
  __bf16* wvb = (__bf16*)(ws + SZ_X + 2 * SZ_W);
  __bf16* wob = (__bf16*)(ws + SZ_X + 3 * SZ_W);
  __bf16* qf  = (__bf16*)(ws + SZ_X + 4 * SZ_W);
  __bf16* kf  = (__bf16*)((char*)qf + SZ_X);
  __bf16* vf  = (__bf16*)((char*)kf + SZ_X);
  __bf16* qhb = (__bf16*)((char*)vf + SZ_X);
  __bf16* khb = (__bf16*)((char*)qhb + SZ_X);
  __bf16* vtb = (__bf16*)((char*)khb + SZ_X);
  __bf16* abf = (__bf16*)((char*)vtb + SZ_X);

  const int nx4 = (SEQ * DIMN) / 4;
  const int nw4 = (DIMN * DIMN) / 4;
  k_cast_bf16<<<(nx4 + 255) / 256, 256, 0, stream>>>(x,  xb,  nx4);
  k_cast_bf16<<<(nw4 + 255) / 256, 256, 0, stream>>>(Wq, wqb, nw4);
  k_cast_bf16<<<(nw4 + 255) / 256, 256, 0, stream>>>(Wk, wkb, nw4);
  k_cast_bf16<<<(nw4 + 255) / 256, 256, 0, stream>>>(Wv, wvb, nw4);
  k_cast_bf16<<<(nw4 + 255) / 256, 256, 0, stream>>>(Wo, wob, nw4);

  dim3 ggrid(DIMN / 256, SEQ / 16);
  k_gemm<false><<<ggrid, 128, 0, stream>>>(xb, wqb, bq, qf, SEQ, DIMN, DIMN);
  k_gemm<false><<<ggrid, 128, 0, stream>>>(xb, wkb, bk, kf, SEQ, DIMN, DIMN);
  k_gemm<false><<<ggrid, 128, 0, stream>>>(xb, wvb, bv, vf, SEQ, DIMN, DIMN);

  k_normrope<<<SEQ, 256, 0, stream>>>(qf, kf, vf, gq, gk, fc, fs, qhb, khb, vtb);

  k_attn<<<dim3(SEQ / 16 / 4, NHEAD), 128, 0, stream>>>(qhb, khb, vtb, abf);

  k_gemm<true><<<ggrid, 128, 0, stream>>>(abf, wob, bo, (float*)d_out, SEQ, DIMN, DIMN);
}
